// BubbleTransformer_35527969473083
// MI455X (gfx1250) — matrix operand order repaired and checked by reading, not compiled, not run on hardware
//
#include <hip/hip_runtime.h>
#include <hip/hip_bf16.h>
#include <stdint.h>
#include <cstddef>
#include <type_traits>

#define B_  4
#define T_  2048
#define D_  1024
#define H_  16
#define DH_ 64
#define L_  2
#define C_  32000
#define TD3 (3 * D_)

typedef __attribute__((ext_vector_type(16))) __bf16 v16bf;
typedef __attribute__((ext_vector_type(8)))  __bf16 v8bf;
typedef __attribute__((ext_vector_type(8)))  float  v8f;

// fp32 -> bf16 (round to nearest even), raw bits
static __device__ __forceinline__ unsigned short f2bf(float f) {
  union { float f; uint32_t u; } v; v.f = f;
  uint32_t u = v.u;
  uint32_t r = (u + 0x7FFFu + ((u >> 16) & 1u)) >> 16;
  return (unsigned short)r;
}

// Assemble a 16-element bf16 fragment from two 8-element (16B) LDS chunks.
static __device__ __forceinline__ v16bf ld_frag(const unsigned short* p0,
                                                const unsigned short* p1) {
  v8bf lo = *reinterpret_cast<const v8bf*>(p0);
  v8bf hi = *reinterpret_cast<const v8bf*>(p1);
  v16bf r;
#pragma unroll
  for (int i = 0; i < 8; ++i) { r[i] = lo[i]; r[i + 8] = hi[i]; }
  return r;
}

// ---------------------------------------------------------------- embedding
__global__ __launch_bounds__(256) void embed_kernel(const int* __restrict__ tokens,
                                                    const float* __restrict__ embW,
                                                    const float* __restrict__ posW,
                                                    float* __restrict__ x) {
  int row = blockIdx.x;              // row = b*T + t
  int t = row & (T_ - 1);
  int tok = tokens[row];
  int d = threadIdx.x * 4;
  float4 e = *reinterpret_cast<const float4*>(embW + (size_t)tok * D_ + d);
  float4 p = *reinterpret_cast<const float4*>(posW + (size_t)t * D_ + d);
  float4 o; o.x = e.x + p.x; o.y = e.y + p.y; o.z = e.z + p.z; o.w = e.w + p.w;
  *reinterpret_cast<float4*>(x + (size_t)row * D_ + d) = o;
}

// ---------------------------------------------------------------- layernorm
__global__ __launch_bounds__(256) void ln_kernel(const float* __restrict__ in,
                                                 const float* __restrict__ g,
                                                 const float* __restrict__ bias,
                                                 float* __restrict__ out,
                                                 long in_row_stride) {
  int row = blockIdx.x;
  int d = threadIdx.x * 4;
  float4 v = *reinterpret_cast<const float4*>(in + (long)row * in_row_stride + d);
  float s = v.x + v.y + v.z + v.w;
  float q = v.x * v.x + v.y * v.y + v.z * v.z + v.w * v.w;
#pragma unroll
  for (int off = 16; off > 0; off >>= 1) {
    s += __shfl_xor(s, off);
    q += __shfl_xor(q, off);
  }
  __shared__ float red[16];
  __shared__ float mu_s, rs_s;
  int wave = threadIdx.x >> 5, lane = threadIdx.x & 31;
  if (lane == 0) { red[wave] = s; red[8 + wave] = q; }
  __syncthreads();
  if (threadIdx.x == 0) {
    float ts = 0.f, tq = 0.f;
#pragma unroll
    for (int i = 0; i < 8; ++i) { ts += red[i]; tq += red[8 + i]; }
    float mu = ts * (1.0f / D_);
    float var = tq * (1.0f / D_) - mu * mu;
    mu_s = mu; rs_s = rsqrtf(var + 1e-5f);
  }
  __syncthreads();
  float mu = mu_s, rs = rs_s;
  float4 gg = *reinterpret_cast<const float4*>(g + d);
  float4 bb = *reinterpret_cast<const float4*>(bias + d);
  float4 o;
  o.x = (v.x - mu) * rs * gg.x + bb.x;
  o.y = (v.y - mu) * rs * gg.y + bb.y;
  o.z = (v.z - mu) * rs * gg.z + bb.z;
  o.w = (v.w - mu) * rs * gg.w + bb.w;
  *reinterpret_cast<float4*>(out + (size_t)row * D_ + d) = o;
}

// ------------------------------------------------- bf16 WMMA GEMM (generic)
// out[M,N] = A[M,K] @ W[K,N] + bias (+ residual). fp32 in/out, bf16 MACs.
// Block tile 128x128, K-step 64, 8 waves = 4(M) x 2(N), wave tile 32x64.
#define GBM 128
#define GBN 128
#define GBK 64
#define GPITCH 72

template <bool HAS_RES, bool GUARD>
__global__ __launch_bounds__(256) void gemm_bf16_kernel(
    const float* __restrict__ A, const float* __restrict__ W,
    const float* __restrict__ bias, const float* __restrict__ res,
    float* __restrict__ out, int M, int N, int K, int Mstore) {
  __shared__ unsigned short Alds[GBM * GPITCH];
  __shared__ unsigned short Blds[GBN * GPITCH];
  const int tid = threadIdx.x;
  const int lane = tid & 31, wave = tid >> 5;
  const int wm = wave >> 1, wn = wave & 1;       // 4 (M) x 2 (N) wave grid
  const int m0 = blockIdx.y * GBM;
  const int n0 = blockIdx.x * GBN;

  v8f acc[2][4];
#pragma unroll
  for (int i = 0; i < 2; ++i)
#pragma unroll
    for (int j = 0; j < 4; ++j) { v8f z = {}; acc[i][j] = z; }

  const int a_row = tid >> 1;              // 0..127
  const int a_col = (tid & 1) * 32;        // 0 / 32
  const int b_kr  = tid >> 2;              // 0..63
  const int b_nb  = (tid & 3) * 32;        // 0/32/64/96

  const int ar  = lane & 15;
  const int akb = (lane >> 4) * 8;         // A: K chunks kb..kb+7 / kb+16..kb+23
  const int bkb = (lane >> 4) * 8;         // B: K chunks kb..kb+7 / kb+16..kb+23

  for (int k0 = 0; k0 < K; k0 += GBK) {
    { // stage A tile (row-major, fp32 -> bf16)
      const float* src = A + (size_t)(m0 + a_row) * K + k0 + a_col;
      unsigned short* dst = &Alds[a_row * GPITCH + a_col];
#pragma unroll
      for (int i = 0; i < 32; i += 4) {
        float4 v = *reinterpret_cast<const float4*>(src + i);
        dst[i]     = f2bf(v.x); dst[i + 1] = f2bf(v.y);
        dst[i + 2] = f2bf(v.z); dst[i + 3] = f2bf(v.w);
      }
      if (k0 + GBK < K) __builtin_prefetch(src + GBK, 0, 1);
    }
    { // stage W tile transposed: Blds[n][k]
      const float* src = W + (size_t)(k0 + b_kr) * N + n0 + b_nb;
#pragma unroll
      for (int i = 0; i < 32; i += 4) {
        float4 v = *reinterpret_cast<const float4*>(src + i);
        Blds[(b_nb + i    ) * GPITCH + b_kr] = f2bf(v.x);
        Blds[(b_nb + i + 1) * GPITCH + b_kr] = f2bf(v.y);
        Blds[(b_nb + i + 2) * GPITCH + b_kr] = f2bf(v.z);
        Blds[(b_nb + i + 3) * GPITCH + b_kr] = f2bf(v.w);
      }
      if (k0 + GBK < K)
        __builtin_prefetch(W + (size_t)(k0 + GBK + b_kr) * N + n0 + b_nb, 0, 1);
    }
    __syncthreads();

#pragma unroll
    for (int kk = 0; kk < 2; ++kk) {       // two K=32 sub-steps per stage
      v16bf af[2];
#pragma unroll
      for (int im = 0; im < 2; ++im) {
        const unsigned short* p =
            &Alds[(wm * 32 + im * 16 + ar) * GPITCH + kk * 32 + akb];
        af[im] = ld_frag(p, p + 16);
      }
      v16bf bfm[4];
#pragma unroll
      for (int in_ = 0; in_ < 4; ++in_) {
        const unsigned short* p =
            &Blds[(wn * 64 + in_ * 16 + ar) * GPITCH + kk * 32 + bkb];
        bfm[in_] = ld_frag(p, p + 16);
      }
#pragma unroll
      for (int im = 0; im < 2; ++im)
#pragma unroll
        for (int in_ = 0; in_ < 4; ++in_)
          acc[im][in_] = __builtin_amdgcn_wmma_f32_16x16x32_bf16(
              false, af[im], false, bfm[in_], (short)0, acc[im][in_], false, false);
    }
    __syncthreads();
  }

  const int mhi = (lane >> 4) * 8;
#pragma unroll
  for (int im = 0; im < 2; ++im) {
#pragma unroll
    for (int in_ = 0; in_ < 4; ++in_) {
      const int col = n0 + wn * 64 + in_ * 16 + (lane & 15);
      const float bv = bias[col];
#pragma unroll
      for (int vr = 0; vr < 8; ++vr) {
        const int row = m0 + wm * 32 + im * 16 + vr + mhi;
        if (!GUARD || row < Mstore) {
          float o = acc[im][in_][vr] + bv;
          if (HAS_RES) o += res[(size_t)row * N + col];
          out[(size_t)row * N + col] = o;
        }
      }
    }
  }
}

// --------------------------------------------------------- flash attention
// qkv: [B*T, 3D] fp32.  out: [B*T, D] fp32.  One block = 128 q rows of (b,h).
#define KPITCH 72
#define VPITCH 40
#define PPITCH 40

__global__ __launch_bounds__(256) void flash_attn_kernel(const float* __restrict__ qkv,
                                                         float* __restrict__ outp) {
  __shared__ unsigned short Klds[32 * KPITCH];       // [klocal][dh] bf16
  __shared__ unsigned short Vt[64 * VPITCH];         // [dh][klocal] bf16
  __shared__ unsigned short Plds[8][16 * PPITCH];    // per-wave P tile

  const int tid = threadIdx.x, lane = tid & 31, wave = tid >> 5;
  const int qb = blockIdx.x, h = blockIdx.y, bb = blockIdx.z;
  const int qrow0 = qb * 128 + wave * 16;
  const size_t rs3 = (size_t)TD3;
  const size_t base = (size_t)bb * T_ * rs3;

  // Q fragments (A layout), scaled by 1/sqrt(DH)=0.125, kept in registers
  v16bf qf[2];
  {
    const int qr = qrow0 + (lane & 15);
    const int kb = (lane >> 4) * 8;
    const float* qp = qkv + base + (size_t)qr * rs3 + h * DH_;
#pragma unroll
    for (int kk = 0; kk < 2; ++kk) {
      v16bf a;
#pragma unroll
      for (int c = 0; c < 2; ++c) {
#pragma unroll
        for (int i = 0; i < 8; i += 4) {
          float4 v = *reinterpret_cast<const float4*>(qp + kk * 32 + kb + c * 16 + i);
          union { unsigned short u; __bf16 b; } t0, t1, t2, t3;
          t0.u = f2bf(v.x * 0.125f); t1.u = f2bf(v.y * 0.125f);
          t2.u = f2bf(v.z * 0.125f); t3.u = f2bf(v.w * 0.125f);
          a[c * 8 + i] = t0.b; a[c * 8 + i + 1] = t1.b;
          a[c * 8 + i + 2] = t2.b; a[c * 8 + i + 3] = t3.b;
        }
      }
      qf[kk] = a;
    }
  }

  v8f o_acc[4];
#pragma unroll
  for (int nt = 0; nt < 4; ++nt) { v8f z = {}; o_acc[nt] = z; }
  float mrow[8], lrow[8];
#pragma unroll
  for (int vr = 0; vr < 8; ++vr) { mrow[vr] = -1e30f; lrow[vr] = 0.0f; }

  const int jmax = qb * 4 + 4;   // causal: k up to (qb+1)*128

  auto tile_body = [&](int j, auto masked_t) {
    constexpr bool MASKED = decltype(masked_t)::value;
    { // cooperative stage K tile and V^T tile
      const int kl = tid >> 3;
      const int cb = (tid & 7) * 8;
      const float* kp = qkv + base + (size_t)(j * 32 + kl) * rs3 + D_     + h * DH_ + cb;
      const float* vp = qkv + base + (size_t)(j * 32 + kl) * rs3 + 2 * D_ + h * DH_ + cb;
#pragma unroll
      for (int i = 0; i < 8; i += 4) {
        float4 kv = *reinterpret_cast<const float4*>(kp + i);
        Klds[kl * KPITCH + cb + i]     = f2bf(kv.x);
        Klds[kl * KPITCH + cb + i + 1] = f2bf(kv.y);
        Klds[kl * KPITCH + cb + i + 2] = f2bf(kv.z);
        Klds[kl * KPITCH + cb + i + 3] = f2bf(kv.w);
        float4 vv = *reinterpret_cast<const float4*>(vp + i);
        Vt[(cb + i    ) * VPITCH + kl] = f2bf(vv.x);
        Vt[(cb + i + 1) * VPITCH + kl] = f2bf(vv.y);
        Vt[(cb + i + 2) * VPITCH + kl] = f2bf(vv.z);
        Vt[(cb + i + 3) * VPITCH + kl] = f2bf(vv.w);
      }
      if (j + 1 < jmax) {                  // pull next K/V slab toward the WGP
        __builtin_prefetch(kp + 32 * rs3, 0, 1);
        __builtin_prefetch(vp + 32 * rs3, 0, 1);
      }
    }
    __syncthreads();

    const int bkb = (lane >> 4) * 8;
    // Batch all K fragment loads, then issue the 4 score WMMAs back-to-back.
    v16bf kf[2][2];
#pragma unroll
    for (int sub = 0; sub < 2; ++sub)
#pragma unroll
      for (int kk = 0; kk < 2; ++kk) {
        const unsigned short* p =
            &Klds[(sub * 16 + (lane & 15)) * KPITCH + kk * 32 + bkb];
        kf[sub][kk] = ld_frag(p, p + 16);
      }
    v8f s01[2];
#pragma unroll
    for (int sub = 0; sub < 2; ++sub) {
      v8f s = {};
      s = __builtin_amdgcn_wmma_f32_16x16x32_bf16(false, qf[0], false, kf[sub][0],
                                                  (short)0, s, false, false);
      s = __builtin_amdgcn_wmma_f32_16x16x32_bf16(false, qf[1], false, kf[sub][1],
                                                  (short)0, s, false, false);
      s01[sub] = s;
    }

    { // online softmax over this 32-wide k slab
      const int mhi = (lane >> 4) * 8;
      float p0[8], p1[8], alpha[8];
#pragma unroll
      for (int vr = 0; vr < 8; ++vr) {
        float s0 = s01[0][vr];
        float s1 = s01[1][vr];
        if (MASKED) {
          const int qrow = qrow0 + vr + mhi;
          const int kc0 = j * 32 + (lane & 15);
          if (kc0 > qrow)      s0 = -1e30f;
          if (kc0 + 16 > qrow) s1 = -1e30f;
        }
        float v = fmaxf(s0, s1);
#pragma unroll
        for (int off = 1; off < 16; off <<= 1) v = fmaxf(v, __shfl_xor(v, off));
        float mn = fmaxf(mrow[vr], v);
        float a = __expf(mrow[vr] - mn);
        float e0 = __expf(s0 - mn), e1 = __expf(s1 - mn);
        float r = e0 + e1;
#pragma unroll
        for (int off = 1; off < 16; off <<= 1) r += __shfl_xor(r, off);
        lrow[vr] = lrow[vr] * a + r;
        mrow[vr] = mn;
        alpha[vr] = a;
        p0[vr] = e0; p1[vr] = e1;
      }
#pragma unroll
      for (int nt = 0; nt < 4; ++nt)
#pragma unroll
        for (int vr = 0; vr < 8; ++vr) o_acc[nt][vr] *= alpha[vr];

      unsigned short* pl = Plds[wave];     // transpose P via per-wave LDS
#pragma unroll
      for (int vr = 0; vr < 8; ++vr) {
        pl[(vr + mhi) * PPITCH + (lane & 15)]      = f2bf(p0[vr]);
        pl[(vr + mhi) * PPITCH + 16 + (lane & 15)] = f2bf(p1[vr]);
      }
    }
    // No block barrier needed: P tile is wave-private and LDS ops from the
    // same wave execute in order (DScnt); compiler inserts s_wait_dscnt.
    { // O += P @ V   (batch all V fragment loads first)
      const int akb = (lane >> 4) * 8;
      const unsigned short* pp = &Plds[wave][(lane & 15) * PPITCH + akb];
      v16bf pf = ld_frag(pp, pp + 16);
      v16bf vfr[4];
#pragma unroll
      for (int nt = 0; nt < 4; ++nt) {
        const unsigned short* vv = &Vt[(nt * 16 + (lane & 15)) * VPITCH + bkb];
        vfr[nt] = ld_frag(vv, vv + 16);
      }
#pragma unroll
      for (int nt = 0; nt < 4; ++nt)
        o_acc[nt] = __builtin_amdgcn_wmma_f32_16x16x32_bf16(
            false, pf, false, vfr[nt], (short)0, o_acc[nt], false, false);
    }
    __syncthreads();
  };

  int j = 0;
  for (; j < qb * 4; ++j)                       // fully-causal-visible tiles
    tile_body(j, std::integral_constant<bool, false>{});
  for (; j < jmax; ++j)                         // 4 diagonal tiles need masking
    tile_body(j, std::integral_constant<bool, true>{});

  const int mhi = (lane >> 4) * 8;
  float inv[8];
#pragma unroll
  for (int vr = 0; vr < 8; ++vr) inv[vr] = 1.0f / lrow[vr];
#pragma unroll
  for (int nt = 0; nt < 4; ++nt)
#pragma unroll
    for (int vr = 0; vr < 8; ++vr) {
      int qrow = qrow0 + vr + mhi;
      outp[((size_t)bb * T_ + qrow) * D_ + h * DH_ + nt * 16 + (lane & 15)] =
          o_acc[nt][vr] * inv[vr];
    }
}

// ------------------------------------------------------------------ driver
extern "C" void kernel_launch(void* const* d_in, const int* in_sizes, int n_in,
                              void* d_out, int out_size, void* d_ws, size_t ws_size,
                              hipStream_t stream) {
  const int*   tokens = (const int*)d_in[0];
  const float* embW   = (const float*)d_in[1];
  const float* posW   = (const float*)d_in[2];
  const float* ln_g   = (const float*)d_in[3];
  const float* ln_b   = (const float*)d_in[4];
  const float* qkv_W  = (const float*)d_in[5];
  const float* qkv_b  = (const float*)d_in[6];
  const float* out_W  = (const float*)d_in[7];
  const float* out_b  = (const float*)d_in[8];
  const float* lnf_g  = (const float*)d_in[9];
  const float* lnf_b  = (const float*)d_in[10];
  const float* head_W = (const float*)d_in[11];
  const float* head_b = (const float*)d_in[12];
  float* logits = (float*)d_out;

  const size_t BT = (size_t)B_ * T_;
  float* ws    = (float*)d_ws;
  float* X     = ws;                         // [BT, D]
  float* LNATT = X + BT * D_;                // [BT, D]  (LN out, reused as attn out)
  float* QKV   = LNATT + BT * D_;            // [BT, 3D]
  float* LNF   = QKV + BT * (size_t)TD3;     // [128, D] zero-padded

  embed_kernel<<<dim3((unsigned)BT), dim3(256), 0, stream>>>(tokens, embW, posW, X);

  for (int l = 0; l < L_; ++l) {
    ln_kernel<<<dim3((unsigned)BT), 256, 0, stream>>>(
        X, ln_g + (size_t)l * D_, ln_b + (size_t)l * D_, LNATT, (long)D_);
    // qkv = LN(x) @ Wqkv + b
    gemm_bf16_kernel<false, false>
        <<<dim3(TD3 / GBN, (unsigned)(BT / GBM)), 256, 0, stream>>>(
            LNATT, qkv_W + (size_t)l * D_ * TD3, qkv_b + (size_t)l * TD3,
            nullptr, QKV, (int)BT, TD3, D_, (int)BT);
    flash_attn_kernel<<<dim3(T_ / 128, H_, B_), 256, 0, stream>>>(QKV, LNATT);
    // x = x + attn @ Wo + b   (fused residual)
    gemm_bf16_kernel<true, false>
        <<<dim3(D_ / GBN, (unsigned)(BT / GBM)), 256, 0, stream>>>(
            LNATT, out_W + (size_t)l * D_ * D_, out_b + (size_t)l * D_,
            X, X, (int)BT, D_, D_, (int)BT);
  }

  hipMemsetAsync(LNF, 0, (size_t)128 * D_ * sizeof(float), stream);
  ln_kernel<<<dim3(B_), 256, 0, stream>>>(
      X + (size_t)(T_ - 1) * D_, lnf_g, lnf_b, LNF, (long)T_ * D_);
  // logits = lnf @ head_W + head_b  (M padded to 128, store-guarded to B rows)
  gemm_bf16_kernel<false, true>
      <<<dim3(C_ / GBN, 1), 256, 0, stream>>>(
          LNF, head_W, head_b, nullptr, logits, 128, C_, D_, B_);
}
